// MultiHeadAttention_70557722739129
// MI455X (gfx1250) — matrix operand order repaired and checked by reading, not compiled, not run on hardware
//
#include <hip/hip_runtime.h>

// ---------------------------------------------------------------------------
// MHA for MI455X (gfx1250, wave32, WMMA).
//   S=4096, dmodel=1024, H=16, dk=dv=64.  ~103 GFLOP total.
// Fused flash-attention (1GB score tensor never touches HBM); all GEMMs via
// v_wmma_f32_16x16x32_f16 with f32 accumulate. K-loops are unrolled x2 with
// ping-pong fragment buffers: prefetch into the idle buffer while WMMAs run
// on the other -> no register-rotation moves, only partial loadcnt waits.
// ---------------------------------------------------------------------------

typedef _Float16 f16;
typedef __attribute__((ext_vector_type(8)))  _Float16 v8h;
typedef __attribute__((ext_vector_type(16))) _Float16 v16h;
typedef __attribute__((ext_vector_type(8)))  float    v8f;

#define S_   4096
#define DM_  1024
#define H_   16
#define DK_  64
#define WPB  8      // waves per block (256 threads)
#define PLD  40     // LDS leading dim (halves) for P tile: 16x32 padded vs bank conflicts

static __device__ __forceinline__ v8f wmma_f16f32(v16h a, v16h b, v8f c) {
  // (neg_a, A, neg_b, B, c_mod, C, reuse_a, reuse_b)
  return __builtin_amdgcn_wmma_f32_16x16x32_f16(false, a, false, b, (short)0, c, false, false);
}

static __device__ __forceinline__ v16h cat16(v8h lo, v8h hi) {
  v16h r;
#pragma unroll
  for (int i = 0; i < 8; ++i) { r[i] = lo[i]; r[i + 8] = hi[i]; }
  return r;
}

// A fragment (16x32 f16) from a row-major matrix, leading dim `ld` halves.
// ISA layout: lane l -> row l&15; K halves {0..7,16..23} (l<16) or {8..15,24..31}.
static __device__ __forceinline__ v16h load_fragA(const f16* p, int ld) {
  int l = threadIdx.x & 31;
  const f16* row = p + (l & 15) * ld + ((l >> 4) * 8);
  v8h lo = *(const v8h*)(row);
  v8h hi = *(const v8h*)(row + 16);
  return cat16(lo, hi);
}

// B fragment (32x16 f16) from B^T stored row-major (rows = columns of B),
// leading dim `ld` halves. Lane l -> column l&15, K halves
// {0..7,16..23} (l<16) or {8..15,24..31} (l>=16), packed 2 halves per VGPR.
static __device__ __forceinline__ v16h load_fragB(const f16* bt, int ld) {
  int l = threadIdx.x & 31;
  const f16* row = bt + (l & 15) * ld + ((l >> 4) * 8);
  v8h lo = *(const v8h*)(row);
  v8h hi = *(const v8h*)(row + 16);
  return cat16(lo, hi);
}

// ---------------------------------------------------------------------------
// Prep: fp32 -> f16 convert / weight transpose (tiny fraction of traffic)
// ---------------------------------------------------------------------------
__global__ __launch_bounds__(256) void cvt_kernel(const float* __restrict__ src,
                                                  f16* __restrict__ dst, int n) {
  int i = blockIdx.x * 256 + threadIdx.x;
  if (i < n) dst[i] = (f16)src[i];
}

// src [H][DM][DK] fp32 -> dst [H][DK][DM] f16  (so B-frags load contiguously)
__global__ __launch_bounds__(256) void transposeW_kernel(const float* __restrict__ src,
                                                         f16* __restrict__ dst) {
  int i = blockIdx.x * 256 + threadIdx.x;     // over H*DM*DK, exact multiple of 256
  int h = i / (DM_ * DK_);
  int r = i % (DM_ * DK_);
  int d = r / DK_, n = r % DK_;
  dst[h * (DK_ * DM_) + n * DM_ + d] = (f16)src[i];
}

// One GEMM pipeline stage: prefetch fragments at kpos into buffer NXT while
// running WMMAs on buffer CUR. CUR/NXT are literal constants at expansion.
#define GEMM_STAGE(kpos, CUR, NXT)                                       \
  do {                                                                   \
    int kn_ = (kpos);                                                    \
    a0[NXT] = load_fragA(xa + kn_, DM_);                                 \
    a1[NXT] = load_fragA(xa + 16 * DM_ + kn_, DM_);                      \
    _Pragma("unroll")                                                    \
    for (int j = 0; j < 4; ++j)                                          \
      b[NXT][j] = load_fragB(wB + (j * 16) * DM_ + kn_, DM_);            \
    _Pragma("unroll")                                                    \
    for (int j = 0; j < 4; ++j) {                                        \
      acc[0][j] = wmma_f16f32(a0[CUR], b[CUR][j], acc[0][j]);            \
      acc[1][j] = wmma_f16f32(a1[CUR], b[CUR][j], acc[1][j]);            \
    }                                                                    \
  } while (0)

// ---------------------------------------------------------------------------
// Per-head projections: qh/kh [H][S][64] row-major, vhT [H][64][S] transposed.
// One wave per (type, head, 32-row tile): M=32, N=64, K=1024.
// ---------------------------------------------------------------------------
__global__ __launch_bounds__(256) void proj_kernel(
    const f16* __restrict__ q16, const f16* __restrict__ k16, const f16* __restrict__ v16p,
    const f16* __restrict__ wqT, const f16* __restrict__ wkT, const f16* __restrict__ wvT,
    f16* __restrict__ qh, f16* __restrict__ kh, f16* __restrict__ vhT) {
  int wave = blockIdx.x * WPB + (threadIdx.x >> 5);
  int t    = wave / (H_ * (S_ / 32));
  int rem  = wave % (H_ * (S_ / 32));
  int h    = rem / (S_ / 32);
  int q0   = (rem % (S_ / 32)) * 32;

  const f16* x  = (t == 0) ? q16 : (t == 1) ? k16 : v16p;
  const f16* xa = x + q0 * DM_;
  const f16* wB = ((t == 0) ? wqT : (t == 1) ? wkT : wvT) + h * (DK_ * DM_);

  v8f acc[2][4] = {};
  v16h a0[2], a1[2], b[2][4];
  a0[0] = load_fragA(xa, DM_);
  a1[0] = load_fragA(xa + 16 * DM_, DM_);
#pragma unroll
  for (int j = 0; j < 4; ++j) b[0][j] = load_fragB(wB + (j * 16) * DM_, DM_);

  for (int kk = 0; kk < DM_; kk += 64) {
    GEMM_STAGE(kk + 32, 0, 1);                 // compute buf0, prefetch buf1
    GEMM_STAGE((kk + 64) & (DM_ - 1), 1, 0);   // compute buf1, prefetch buf0
  }

  int l = threadIdx.x & 31;
  int ncol = l & 15;
  if (t == 2) {
    f16* oT = vhT + h * (DK_ * S_);
#pragma unroll
    for (int m = 0; m < 2; ++m) {
      int mbase = q0 + m * 16 + ((l >> 4) * 8);
#pragma unroll
      for (int j = 0; j < 4; ++j)
#pragma unroll
        for (int i = 0; i < 8; ++i)
          oT[(j * 16 + ncol) * S_ + (mbase + i)] = (f16)acc[m][j][i];
    }
  } else {
    f16* o = ((t == 0) ? qh : kh) + h * (S_ * DK_);
#pragma unroll
    for (int m = 0; m < 2; ++m) {
      int mbase = q0 + m * 16 + ((l >> 4) * 8);
#pragma unroll
      for (int j = 0; j < 4; ++j)
#pragma unroll
        for (int i = 0; i < 8; ++i)
          o[(mbase + i) * DK_ + (j * 16 + ncol)] = (f16)acc[m][j][i];
    }
  }
}

// One flash-attention stage over a 32-key block. Consumes K/V fragment buffer
// CUR, prefetches buffer NXT for key block at KBN. CUR/NXT are literals.
// LDS P-tile reuse across stages is safe: LDS ops are in-order per wave.
#define ATTN_BODY(KBN, CUR, NXT)                                              \
  do {                                                                        \
    v8f z = {};                                                               \
    v8f s0 = wmma_f16f32(qa1, kf[CUR][1], wmma_f16f32(qa0, kf[CUR][0], z));   \
    v8f s1 = wmma_f16f32(qa1, kf[CUR][3], wmma_f16f32(qa0, kf[CUR][2], z));   \
    kf[NXT][0] = load_fragB(khh + (KBN) * DK_ + 0, DK_);                      \
    kf[NXT][1] = load_fragB(khh + (KBN) * DK_ + 32, DK_);                     \
    kf[NXT][2] = load_fragB(khh + ((KBN) + 16) * DK_ + 0, DK_);               \
    kf[NXT][3] = load_fragB(khh + ((KBN) + 16) * DK_ + 32, DK_);              \
    _Pragma("unroll")                                                         \
    for (int f = 0; f < 4; ++f)                                               \
      vb[NXT][f] = load_fragB(vT + (f * 16) * S_ + (KBN), S_);                \
    float alpha[8];                                                           \
    _Pragma("unroll")                                                         \
    for (int i = 0; i < 8; ++i) {                                             \
      float mx = fmaxf(s0[i], s1[i]);                                         \
      mx = fmaxf(mx, __shfl_xor(mx, 1, 16));                                  \
      mx = fmaxf(mx, __shfl_xor(mx, 2, 16));                                  \
      mx = fmaxf(mx, __shfl_xor(mx, 4, 16));                                  \
      mx = fmaxf(mx, __shfl_xor(mx, 8, 16));                                  \
      float mnew = fmaxf(mrow[i], mx);                                        \
      alpha[i] = __builtin_amdgcn_exp2f((mrow[i] - mnew) * c);                \
      mrow[i] = mnew;                                                         \
      float p0 = __builtin_amdgcn_exp2f((s0[i] - mnew) * c);                  \
      float p1 = __builtin_amdgcn_exp2f((s1[i] - mnew) * c);                  \
      float rs = p0 + p1;                                                     \
      rs += __shfl_xor(rs, 1, 16);                                            \
      rs += __shfl_xor(rs, 2, 16);                                            \
      rs += __shfl_xor(rs, 4, 16);                                            \
      rs += __shfl_xor(rs, 8, 16);                                            \
      lrow[i] = lrow[i] * alpha[i] + rs;                                      \
      Pw[(i + 8 * half) * PLD + col]      = (f16)p0;                          \
      Pw[(i + 8 * half) * PLD + 16 + col] = (f16)p1;                          \
    }                                                                         \
    _Pragma("unroll")                                                         \
    for (int f = 0; f < 4; ++f)                                               \
      _Pragma("unroll")                                                       \
      for (int i = 0; i < 8; ++i) acc[f][i] *= alpha[i];                      \
    v16h pa = load_fragA(Pw, PLD);                                            \
    _Pragma("unroll")                                                         \
    for (int f = 0; f < 4; ++f) acc[f] = wmma_f16f32(pa, vb[CUR][f], acc[f]); \
  } while (0)

// ---------------------------------------------------------------------------
// Flash attention: one wave per (head, 16-query tile). Online softmax over
// 32-key blocks; softmax VALU work hides the next block's K/V load latency.
// ---------------------------------------------------------------------------
__global__ __launch_bounds__(256) void attn_kernel(
    const f16* __restrict__ qh, const f16* __restrict__ kh,
    const f16* __restrict__ vhT, f16* __restrict__ hc) {
  __shared__ _Float16 Plds[WPB][16 * PLD];

  int wid  = threadIdx.x >> 5;
  int wave = blockIdx.x * WPB + wid;
  int h    = wave / (S_ / 16);
  int q0   = (wave % (S_ / 16)) * 16;

  const f16* qhh = qh + h * (S_ * DK_);
  const f16* khh = kh + h * (S_ * DK_);
  const f16* vT  = vhT + h * (DK_ * S_);

  v16h qa0 = load_fragA(qhh + q0 * DK_ + 0, DK_);
  v16h qa1 = load_fragA(qhh + q0 * DK_ + 32, DK_);

  const float c = 0.18033688011112042f;  // log2(e) / sqrt(dk): scale folded into exp2
  float mrow[8], lrow[8];
#pragma unroll
  for (int i = 0; i < 8; ++i) { mrow[i] = -1e30f; lrow[i] = 0.0f; }
  v8f acc[4] = {};

  int l = threadIdx.x & 31;
  int half = l >> 4, col = l & 15;
  _Float16* Pw = &Plds[wid][0];

  // preload first block's K and V fragments into buffer 0
  v16h kf[2][4], vb[2][4];
  kf[0][0] = load_fragB(khh + 0, DK_);
  kf[0][1] = load_fragB(khh + 32, DK_);
  kf[0][2] = load_fragB(khh + 16 * DK_ + 0, DK_);
  kf[0][3] = load_fragB(khh + 16 * DK_ + 32, DK_);
#pragma unroll
  for (int f = 0; f < 4; ++f) vb[0][f] = load_fragB(vT + (f * 16) * S_, S_);

  for (int kb = 0; kb < S_; kb += 64) {
    ATTN_BODY(kb + 32, 0, 1);                  // keys kb..kb+31, prefetch buf1
    ATTN_BODY((kb + 64) & (S_ - 1), 1, 0);     // keys kb+32..kb+63, prefetch buf0
  }

#pragma unroll
  for (int i = 0; i < 8; ++i) lrow[i] = 1.0f / lrow[i];
  int mbase = q0 + 8 * half;
#pragma unroll
  for (int f = 0; f < 4; ++f)
#pragma unroll
    for (int i = 0; i < 8; ++i)
      hc[(mbase + i) * DM_ + h * DK_ + f * 16 + col] = (f16)(acc[f][i] * lrow[i]);
}

// ---------------------------------------------------------------------------
// Output projection: Y = head_cat @ Wo^T. Wo [out][in] row-major == B^T.
// One wave per 32x64 output tile, K=1024, ping-pong double-buffered.
// ---------------------------------------------------------------------------
__global__ __launch_bounds__(256) void outproj_kernel(
    const f16* __restrict__ hc, const f16* __restrict__ wo16,
    float* __restrict__ out) {
  int wave   = blockIdx.x * WPB + (threadIdx.x >> 5);
  int nstrip = wave / (S_ / 32);
  int q0     = (wave % (S_ / 32)) * 32;
  int n0     = nstrip * 64;

  const f16* xa = hc + q0 * DM_;
  const f16* wB = wo16 + n0 * DM_;

  v8f acc[2][4] = {};
  v16h a0[2], a1[2], b[2][4];
  a0[0] = load_fragA(xa, DM_);
  a1[0] = load_fragA(xa + 16 * DM_, DM_);
#pragma unroll
  for (int j = 0; j < 4; ++j) b[0][j] = load_fragB(wB + (j * 16) * DM_, DM_);

  for (int kk = 0; kk < DM_; kk += 64) {
    GEMM_STAGE(kk + 32, 0, 1);
    GEMM_STAGE((kk + 64) & (DM_ - 1), 1, 0);
  }

  int l = threadIdx.x & 31;
  int col = l & 15;
#pragma unroll
  for (int m = 0; m < 2; ++m) {
    int mbase = q0 + m * 16 + 8 * (l >> 4);
#pragma unroll
    for (int j = 0; j < 4; ++j)
#pragma unroll
      for (int i = 0; i < 8; ++i)
        out[(mbase + i) * DM_ + n0 + j * 16 + col] = acc[m][j][i];
  }
}

// ---------------------------------------------------------------------------
extern "C" void kernel_launch(void* const* d_in, const int* in_sizes, int n_in,
                              void* d_out, int out_size, void* d_ws, size_t ws_size,
                              hipStream_t stream) {
  (void)in_sizes; (void)n_in; (void)out_size; (void)ws_size;
  const float* q  = (const float*)d_in[0];
  const float* k  = (const float*)d_in[1];
  const float* v  = (const float*)d_in[2];
  const float* Wq = (const float*)d_in[3];
  const float* Wk = (const float*)d_in[4];
  const float* Wv = (const float*)d_in[5];
  const float* Wo = (const float*)d_in[6];
  float* out = (float*)d_out;

  char* ws = (char*)d_ws;                      // 64 MB total staging, all f16
  f16* q16  = (f16*)(ws + (size_t)(0 << 20));  // 8 MB  [S][DM]
  f16* k16  = (f16*)(ws + (size_t)(8 << 20));  // 8 MB
  f16* v16p = (f16*)(ws + (size_t)(16 << 20)); // 8 MB
  f16* wqT  = (f16*)(ws + (size_t)(24 << 20)); // 2 MB  [H][DK][DM]
  f16* wkT  = (f16*)(ws + (size_t)(26 << 20)); // 2 MB
  f16* wvT  = (f16*)(ws + (size_t)(28 << 20)); // 2 MB
  f16* wo16 = (f16*)(ws + (size_t)(30 << 20)); // 2 MB  [1024][1024]
  f16* qh   = (f16*)(ws + (size_t)(32 << 20)); // 8 MB  [H][S][DK]
  f16* kh   = (f16*)(ws + (size_t)(40 << 20)); // 8 MB  [H][S][DK]
  f16* vhT  = (f16*)(ws + (size_t)(48 << 20)); // 8 MB  [H][DK][S]
  f16* hc   = (f16*)(ws + (size_t)(56 << 20)); // 8 MB  [S][DM]

  int nqkv = S_ * DM_;            // 4,194,304
  int nw   = H_ * DM_ * DK_;      // 1,048,576 (== 1024*1024 for Wo)
  cvt_kernel<<<nqkv / 256, 256, 0, stream>>>(q, q16, nqkv);
  cvt_kernel<<<nqkv / 256, 256, 0, stream>>>(k, k16, nqkv);
  cvt_kernel<<<nqkv / 256, 256, 0, stream>>>(v, v16p, nqkv);
  cvt_kernel<<<nw / 256, 256, 0, stream>>>(Wo, wo16, nw);
  transposeW_kernel<<<nw / 256, 256, 0, stream>>>(Wq, wqT);
  transposeW_kernel<<<nw / 256, 256, 0, stream>>>(Wk, wkT);
  transposeW_kernel<<<nw / 256, 256, 0, stream>>>(Wv, wvT);

  proj_kernel<<<(3 * H_ * (S_ / 32)) / WPB, 256, 0, stream>>>(
      q16, k16, v16p, wqT, wkT, wvT, qh, kh, vhT);
  attn_kernel<<<(H_ * (S_ / 16)) / WPB, 256, 0, stream>>>(qh, kh, vhT, hc);
  outproj_kernel<<<((S_ / 32) * (DM_ / 64)) / WPB, 256, 0, stream>>>(hc, wo16, out);
}
